// Attention_51299089383683
// MI455X (gfx1250) — matrix operand order repaired and checked by reading, not compiled, not run on hardware
//
#include <hip/hip_runtime.h>

// ---------------------------------------------------------------------------
// Multi-head attention, B=2 N=2048 C=1024 H=16 D=64 on MI455X (gfx1250).
// All matmuls via v_wmma_f32_16x16x32_f16 (wave32). Flash-attention middle
// stage (score matrix never hits HBM). Double-buffered LDS staging with
// sched_barrier fences so global loads stay in flight under the WMMA pipe;
// f16 tiles (flash K, proj A) fetched by the Tensor Data Mover.
// ---------------------------------------------------------------------------

#define EMB   1024
#define N3    3072
#define SEQ   2048
#define HEADS 16
#define HD    64

typedef __attribute__((ext_vector_type(16))) _Float16 v16h;
typedef __attribute__((ext_vector_type(8)))  _Float16 v8h;
typedef __attribute__((ext_vector_type(4)))  _Float16 v4h;
typedef __attribute__((ext_vector_type(8)))  float    v8f;

#if defined(__HIP_DEVICE_COMPILE__) && __has_builtin(__builtin_amdgcn_tensor_load_to_lds)
#define HAVE_TDM 1
typedef __attribute__((ext_vector_type(4))) unsigned int u32x4;
typedef __attribute__((ext_vector_type(8))) int          i32x8;
typedef __attribute__((ext_vector_type(4))) int          i32x4;
#else
#define HAVE_TDM 0
#endif

#if defined(__HIP_DEVICE_COMPILE__) && __has_builtin(__builtin_amdgcn_sched_barrier)
#define SCHED_FENCE() __builtin_amdgcn_sched_barrier(0)
#else
#define SCHED_FENCE() ((void)0)
#endif

// Build a 16-element f16 WMMA fragment from two 16-byte (b128) loads.
static __device__ __forceinline__ v16h ld_frag(const _Float16* lo,
                                               const _Float16* hi) {
  v8h a = *(const v8h*)lo;
  v8h b = *(const v8h*)hi;
  return __builtin_shufflevector(a, b, 0, 1, 2, 3, 4, 5, 6, 7,
                                       8, 9, 10, 11, 12, 13, 14, 15);
}

static __device__ __forceinline__ v8f wmma16(v16h a, v16h b, v8f c) {
  return __builtin_amdgcn_wmma_f32_16x16x32_f16(
      /*neg_a=*/false, a, /*neg_b=*/false, b,
      /*c_mod=*/(short)0, c, /*reuse_a=*/false, /*reuse_b=*/false);
}

#if HAVE_TDM
// TDM 2D f16 tile load (ISA 8.3/8.4). Tile (tile_d0 x tile_d1) elems from a
// row-major tensor (row length dim0, dim1 rows, row stride stride0), written
// contiguously to LDS. 6-arg builtin form on this toolchain.
static __device__ __forceinline__ void tdm_load_2d_f16(
    const _Float16* gsrc, _Float16* ldst, unsigned tile_d0, unsigned tile_d1,
    unsigned dim0, unsigned dim1, unsigned stride0) {
  unsigned long long ga = (unsigned long long)(size_t)gsrc;
  unsigned int lds = (unsigned int)(size_t)(void*)ldst;
  u32x4 g0;
  g0[0] = 1u;                                            // count=1, user D#
  g0[1] = lds;                                           // lds_addr
  g0[2] = (unsigned int)(ga & 0xffffffffu);              // global_addr[31:0]
  g0[3] = (unsigned int)((ga >> 32) & 0x01ffffffu)       // global_addr[56:32]
          | 0x80000000u;                                 // type = 2 ("image")
  i32x8 g1;
  g1[0] = 0x00010000;                        // wg_mask=0, data_size=1 (2B)
  g1[1] = (int)(dim0 << 16);                 // tensor_dim0 lo
  g1[2] = (int)((dim0 >> 16) | (dim1 << 16));  // dim0 hi | tensor_dim1 lo
  g1[3] = (int)((dim1 >> 16) | (tile_d0 << 16));  // dim1 hi | tile_dim0
  g1[4] = (int)tile_d1;                      // tile_dim1, tile_dim2=0
  g1[5] = (int)stride0;                      // tensor_dim0_stride lo
  g1[6] = 0;                                 // stride0 hi | dim1_stride lo
  g1[7] = 0;
  i32x4 z4 = {0, 0, 0, 0};                   // groups 2/3 unused (2D)
  i32x8 z8 = {0, 0, 0, 0, 0, 0, 0, 0};
  __builtin_amdgcn_tensor_load_to_lds(g0, g1, z4, z4, z8, 0);
}
#endif

// ---------------------------------------------------------------------------
// Kernel 1: qkv = x @ w_qkv + b_qkv ; scatter to Q(scaled)/K/V [B,H,N,D] f16.
// 128x128 tile, K-step 32, double-buffered LDS; 8 waves x (2x4) WMMA tiles.
// ---------------------------------------------------------------------------
__global__ __launch_bounds__(256) void qkv_gemm_kernel(
    const float* __restrict__ x,     // [4096,1024]
    const float* __restrict__ w,     // [1024,3072]
    const float* __restrict__ bias,  // [3072]
    _Float16* __restrict__ qb, _Float16* __restrict__ kb,
    _Float16* __restrict__ vb) {
  __shared__ __align__(16) _Float16 At[2][128 * 32];  // [m][k]
  __shared__ __align__(16) _Float16 Bt[2][128 * 32];  // [n][k]

  const int tid  = threadIdx.x;
  const int lane = tid & 31, wave = tid >> 5;
  const int l16  = lane & 15, half = lane >> 4;
  const int wm   = wave & 3,  wn   = wave >> 2;
  const int m0   = blockIdx.y * 128, n0 = blockIdx.x * 128;

  v8f acc[2][4] = {};
  float4 fa[4], fb[4];

  auto load_tiles = [&](int k0) {  // global -> regs (all loads in flight)
#pragma unroll
    for (int i = 0; i < 4; ++i) {
      int e = (i * 256 + tid) * 4;
      fa[i] = *(const float4*)(x + (size_t)(m0 + (e >> 5)) * EMB + k0 + (e & 31));
      fb[i] = *(const float4*)(w + (size_t)(k0 + (e >> 7)) * N3 + n0 + (e & 127));
    }
  };
  auto store_tiles = [&](int buf) {  // regs -> LDS (f32->f16)
#pragma unroll
    for (int i = 0; i < 4; ++i) {
      int e = (i * 256 + tid) * 4;
      int r = e >> 5, c = e & 31;
      v4h t;
      t[0] = (_Float16)fa[i].x; t[1] = (_Float16)fa[i].y;
      t[2] = (_Float16)fa[i].z; t[3] = (_Float16)fa[i].w;
      *(v4h*)(&At[buf][r * 32 + c]) = t;
      int kk = e >> 7, n = e & 127;
      Bt[buf][(n + 0) * 32 + kk] = (_Float16)fb[i].x;
      Bt[buf][(n + 1) * 32 + kk] = (_Float16)fb[i].y;
      Bt[buf][(n + 2) * 32 + kk] = (_Float16)fb[i].z;
      Bt[buf][(n + 3) * 32 + kk] = (_Float16)fb[i].w;
    }
  };

  load_tiles(0);
  store_tiles(0);
  for (int it = 0; it < EMB / 32; ++it) {
    const int cur = it & 1;
    const bool more = (it + 1) < EMB / 32;
    __syncthreads();
    if (more) load_tiles((it + 1) * 32);  // stay in flight under WMMAs
    SCHED_FENCE();

    v16h af[2];
#pragma unroll
    for (int mt = 0; mt < 2; ++mt) {
      const _Float16* p = &At[cur][(wm * 32 + mt * 16 + l16) * 32];
      af[mt] = ld_frag(p + half * 16, p + half * 16 + 8);
    }
#pragma unroll
    for (int nt = 0; nt < 4; ++nt) {
      const _Float16* p = &Bt[cur][(wn * 64 + nt * 16 + l16) * 32 + half * 16];
      v16h bfr = ld_frag(p, p + 8);
#pragma unroll
      for (int mt = 0; mt < 2; ++mt)
        acc[mt][nt] = wmma16(af[mt], bfr, acc[mt][nt]);
    }
    SCHED_FENCE();
    if (more) store_tiles(cur ^ 1);
  }

  // Epilogue: + bias, Q *= D^-0.5, scatter to [B,H,N,D].
  const float scale = 0.125f;  // 64^-0.5
#pragma unroll
  for (int mt = 0; mt < 2; ++mt)
#pragma unroll
    for (int nt = 0; nt < 4; ++nt) {
      int n = n0 + wn * 64 + nt * 16 + l16;
      float bn = bias[n];
      int which = n >> 10, cc = n & 1023;
      int hh = cc >> 6, d = cc & 63;
#pragma unroll
      for (int r = 0; r < 8; ++r) {
        int m = m0 + wm * 32 + mt * 16 + r + 8 * half;
        int bidx = m >> 11, tok = m & 2047;
        size_t dst = (((size_t)(bidx * HEADS + hh)) * SEQ + tok) * HD + d;
        float v = acc[mt][nt][r] + bn;
        if (which == 0)      qb[dst] = (_Float16)(v * scale);
        else if (which == 1) kb[dst] = (_Float16)v;
        else                 vb[dst] = (_Float16)v;
      }
    }
}

// ---------------------------------------------------------------------------
// Kernel 2: flash attention. One block per (b,h,128-row q-block).
// K-tiles via Tensor Data Mover (wave 0), V transposed through registers,
// both double-buffered so DMA/loads overlap the WMMA + softmax work.
// ---------------------------------------------------------------------------
__global__ __launch_bounds__(256) void flash_attn_kernel(
    const _Float16* __restrict__ Q, const _Float16* __restrict__ K,
    const _Float16* __restrict__ V, _Float16* __restrict__ O) {
  __shared__ __align__(16) _Float16 Kt[2][64 * 64];     // [key][d]
  __shared__ __align__(16) _Float16 Vt[2][64 * 64];     // [d][key]
  __shared__ __align__(16) _Float16 Pt[8 * 16 * 64];    // per-wave P scratch

  const int tid  = threadIdx.x;
  const int lane = tid & 31, wave = tid >> 5;
  const int l16  = lane & 15, half = lane >> 4;
  const int bid  = blockIdx.x;
  const int qblk = bid & 15;
  const int bh   = bid >> 4;  // b*16+h
  const size_t base = (size_t)bh * SEQ * HD;

  const _Float16* Qb = Q + base;
  const _Float16* Kb = K + base;
  const _Float16* Vb = V + base;

  // Q fragments (A-matrix) for this wave's 16 rows, D=64 -> 2 K-chunks.
  const int qrow = qblk * 128 + wave * 16 + l16;
  v16h qf[2];
#pragma unroll
  for (int kc = 0; kc < 2; ++kc) {
    const _Float16* p = Qb + (size_t)qrow * HD + kc * 32;
    qf[kc] = ld_frag(p + half * 16, p + half * 16 + 8);
  }

  v8h rv[2];
#if !HAVE_TDM
  v8h rk[2];
#endif
  auto load_kv = [&](int j) {
    const _Float16* Vj = Vb + (size_t)j * 64 * HD;
#pragma unroll
    for (int i = 0; i < 2; ++i) {
      int e = (i * 256 + tid) * 8;
      rv[i] = *(const v8h*)(Vj + e);
#if !HAVE_TDM
      rk[i] = *(const v8h*)(Kb + (size_t)j * 64 * HD + e);
#endif
    }
  };
  auto store_kv = [&](int buf) {
#pragma unroll
    for (int i = 0; i < 2; ++i) {
      int e = (i * 256 + tid) * 8;
      int key = e >> 6, d0 = e & 63;
#pragma unroll
      for (int q = 0; q < 8; ++q) Vt[buf][(d0 + q) * 64 + key] = rv[i][q];
#if !HAVE_TDM
      *(v8h*)(&Kt[buf][e]) = rk[i];
#endif
    }
  };

  float rowmax[8], rowsum[8];
  v8f oacc[4] = {};
#pragma unroll
  for (int r = 0; r < 8; ++r) { rowmax[r] = -3.0e38f; rowsum[r] = 0.f; }

  // Prologue: stage block 0.
  load_kv(0);
#if HAVE_TDM
  if (wave == 0) tdm_load_2d_f16(Kb, &Kt[0][0], HD, 64, HD, SEQ, HD);
#endif
  store_kv(0);
#if HAVE_TDM
  if (wave == 0) __builtin_amdgcn_s_wait_tensorcnt(0);
#endif
  __syncthreads();

  for (int j = 0; j < SEQ / 64; ++j) {
    const int cur = j & 1;
    const bool more = (j + 1) < SEQ / 64;
    if (more) {
      load_kv(j + 1);  // global loads in flight during compute
#if HAVE_TDM
      if (wave == 0)
        tdm_load_2d_f16(Kb + (size_t)(j + 1) * 64 * HD, &Kt[cur ^ 1][0],
                        HD, 64, HD, SEQ, HD);
#endif
    }
    SCHED_FENCE();

    // S = Q*scale @ K^T for 64 keys: 4 N-tiles x 2 K-chunks of WMMA.
    v8f s[4] = {};
#pragma unroll
    for (int nt = 0; nt < 4; ++nt)
#pragma unroll
      for (int kc = 0; kc < 2; ++kc) {
        const _Float16* p =
            &Kt[cur][(nt * 16 + l16) * HD + kc * 32 + half * 16];
        v16h kf = ld_frag(p, p + 8);
        s[nt] = wmma16(qf[kc], kf, s[nt]);
      }

    // Online softmax (row stats shared across 16-lane groups via shfl_xor).
    _Float16* Pw = Pt + wave * 16 * 64;
#pragma unroll
    for (int r = 0; r < 8; ++r) {
      float m = fmaxf(fmaxf(s[0][r], s[1][r]), fmaxf(s[2][r], s[3][r]));
#pragma unroll
      for (int off = 1; off < 16; off <<= 1)
        m = fmaxf(m, __shfl_xor(m, off, 32));
      float mnew = fmaxf(rowmax[r], m);
      float corr = __expf(rowmax[r] - mnew);
      rowmax[r] = mnew;
      float psum = 0.f;
#pragma unroll
      for (int nt = 0; nt < 4; ++nt) {
        float p = __expf(s[nt][r] - mnew);
        s[nt][r] = p;
        psum += p;
      }
#pragma unroll
      for (int off = 1; off < 16; off <<= 1)
        psum += __shfl_xor(psum, off, 32);
      rowsum[r] = rowsum[r] * corr + psum;
#pragma unroll
      for (int nt = 0; nt < 4; ++nt) oacc[nt][r] *= corr;
      int mrow = r + 8 * half;  // C-matrix row this lane holds
#pragma unroll
      for (int nt = 0; nt < 4; ++nt)
        Pw[mrow * 64 + nt * 16 + l16] = (_Float16)s[nt][r];
    }
    // Wave-local LDS RAW: CDNA5 split dependency counter wait.
    asm volatile("s_wait_dscnt 0x0" ::: "memory");

    // O += P @ V  (P as A-matrix from LDS, V^T tile as B-matrix).
    v16h pf[2];
#pragma unroll
    for (int kc = 0; kc < 2; ++kc) {
      const _Float16* p = Pw + l16 * 64 + kc * 32;
      pf[kc] = ld_frag(p + half * 16, p + half * 16 + 8);
    }
#pragma unroll
    for (int nt = 0; nt < 4; ++nt)
#pragma unroll
      for (int kc = 0; kc < 2; ++kc) {
        const _Float16* p =
            &Vt[cur][(nt * 16 + l16) * 64 + kc * 32 + half * 16];
        v16h vf = ld_frag(p, p + 8);
        oacc[nt] = wmma16(pf[kc], vf, oacc[nt]);
      }
    SCHED_FENCE();

    if (more) {
      store_kv(cur ^ 1);
#if HAVE_TDM
      if (wave == 0) __builtin_amdgcn_s_wait_tensorcnt(0);
#endif
    }
    __syncthreads();
  }

  // Normalize and write attention output to [B*SEQ, EMB] f16 workspace.
  const int h = bh & 15, b = bh >> 4;
#pragma unroll
  for (int r = 0; r < 8; ++r) {
    float inv = 1.0f / rowsum[r];
    int tok = qblk * 128 + wave * 16 + r + 8 * half;
    size_t off = ((size_t)b * SEQ + tok) * EMB + h * HD;
#pragma unroll
    for (int nt = 0; nt < 4; ++nt)
      O[off + nt * 16 + l16] = (_Float16)(oacc[nt][r] * inv);
  }
}

// ---------------------------------------------------------------------------
// Kernel 3: out = attn @ w_proj + b_proj  (f16 A via TDM, f32 B->f16).
// ---------------------------------------------------------------------------
__global__ __launch_bounds__(256) void proj_gemm_kernel(
    const _Float16* __restrict__ a,  // [4096,1024] f16
    const float* __restrict__ w,     // [1024,1024]
    const float* __restrict__ bias,  // [1024]
    float* __restrict__ out) {       // [4096,1024]
  __shared__ __align__(16) _Float16 At[2][128 * 32];
  __shared__ __align__(16) _Float16 Bt[2][128 * 32];

  const int tid  = threadIdx.x;
  const int lane = tid & 31, wave = tid >> 5;
  const int l16  = lane & 15, half = lane >> 4;
  const int wm   = wave & 3,  wn   = wave >> 2;
  const int m0   = blockIdx.y * 128, n0 = blockIdx.x * 128;

  v8f acc[2][4] = {};
  float4 fb[4];
#if !HAVE_TDM
  v8h ra[2];
#endif

  auto load_tiles = [&](int k0) {
#pragma unroll
    for (int i = 0; i < 4; ++i) {
      int e = (i * 256 + tid) * 4;
      fb[i] = *(const float4*)(w + (size_t)(k0 + (e >> 7)) * EMB + n0 + (e & 127));
    }
#if !HAVE_TDM
#pragma unroll
    for (int i = 0; i < 2; ++i) {
      int e = (i * 256 + tid) * 8;
      ra[i] = *(const v8h*)(a + (size_t)(m0 + (e >> 5)) * EMB + k0 + (e & 31));
    }
#endif
  };
  auto store_tiles = [&](int buf) {
#pragma unroll
    for (int i = 0; i < 4; ++i) {
      int e = (i * 256 + tid) * 4;
      int kk = e >> 7, n = e & 127;
      Bt[buf][(n + 0) * 32 + kk] = (_Float16)fb[i].x;
      Bt[buf][(n + 1) * 32 + kk] = (_Float16)fb[i].y;
      Bt[buf][(n + 2) * 32 + kk] = (_Float16)fb[i].z;
      Bt[buf][(n + 3) * 32 + kk] = (_Float16)fb[i].w;
    }
#if !HAVE_TDM
#pragma unroll
    for (int i = 0; i < 2; ++i) {
      int e = (i * 256 + tid) * 8;
      *(v8h*)(&At[buf][(e >> 5) * 32 + (e & 31)]) = ra[i];
    }
#endif
  };

  // Prologue: stage k0 = 0.
  load_tiles(0);
#if HAVE_TDM
  // A tile = 128 rows x 32 cols of row-major [4096,1024] f16 -> LDS [m][k].
  if (wave == 0)
    tdm_load_2d_f16(a + (size_t)m0 * EMB, &At[0][0], 32, 128, EMB, 4096, EMB);
#endif
  store_tiles(0);
#if HAVE_TDM
  if (wave == 0) __builtin_amdgcn_s_wait_tensorcnt(0);
#endif

  for (int it = 0; it < EMB / 32; ++it) {
    const int cur = it & 1;
    const bool more = (it + 1) < EMB / 32;
    __syncthreads();
    if (more) {
      load_tiles((it + 1) * 32);
#if HAVE_TDM
      if (wave == 0)
        tdm_load_2d_f16(a + (size_t)m0 * EMB + (it + 1) * 32, &At[cur ^ 1][0],
                        32, 128, EMB, 4096, EMB);
#endif
    }
    SCHED_FENCE();

    v16h af[2];
#pragma unroll
    for (int mt = 0; mt < 2; ++mt) {
      const _Float16* p = &At[cur][(wm * 32 + mt * 16 + l16) * 32];
      af[mt] = ld_frag(p + half * 16, p + half * 16 + 8);
    }
#pragma unroll
    for (int nt = 0; nt < 4; ++nt) {
      const _Float16* p = &Bt[cur][(wn * 64 + nt * 16 + l16) * 32 + half * 16];
      v16h bfr = ld_frag(p, p + 8);
#pragma unroll
      for (int mt = 0; mt < 2; ++mt)
        acc[mt][nt] = wmma16(af[mt], bfr, acc[mt][nt]);
    }
    SCHED_FENCE();
    if (more) {
      store_tiles(cur ^ 1);
#if HAVE_TDM
      if (wave == 0) __builtin_amdgcn_s_wait_tensorcnt(0);
#endif
    }
  }

#pragma unroll
  for (int mt = 0; mt < 2; ++mt)
#pragma unroll
    for (int nt = 0; nt < 4; ++nt) {
      int n = n0 + wn * 64 + nt * 16 + l16;
      float bn = bias[n];
#pragma unroll
      for (int r = 0; r < 8; ++r) {
        int m = m0 + wm * 32 + mt * 16 + r + 8 * half;
        out[(size_t)m * EMB + n] = acc[mt][nt][r] + bn;
      }
    }
}

// ---------------------------------------------------------------------------
extern "C" void kernel_launch(void* const* d_in, const int* in_sizes, int n_in,
                              void* d_out, int out_size, void* d_ws,
                              size_t ws_size, hipStream_t stream) {
  const float* x      = (const float*)d_in[0];
  const float* w_qkv  = (const float*)d_in[1];
  const float* b_qkv  = (const float*)d_in[2];
  const float* w_proj = (const float*)d_in[3];
  const float* b_proj = (const float*)d_in[4];
  float* out = (float*)d_out;

  const size_t PER = (size_t)2 * HEADS * SEQ * HD;  // 4M f16 elements
  _Float16* qb = (_Float16*)d_ws;
  _Float16* kb = qb + PER;
  _Float16* vb = kb + PER;
  _Float16* ab = vb + PER;  // attention output [4096,1024] f16

  qkv_gemm_kernel<<<dim3(24, 32), 256, 0, stream>>>(x, w_qkv, b_qkv,
                                                    qb, kb, vb);
  flash_attn_kernel<<<dim3(512), 256, 0, stream>>>(qb, kb, vb, ab);
  proj_gemm_kernel<<<dim3(8, 32), 256, 0, stream>>>(ab, w_proj, b_proj, out);
}
